// Demo_38147899523261
// MI455X (gfx1250) — matrix operand order repaired and checked by reading, not compiled, not run on hardware
//
#include <hip/hip_runtime.h>
#include <hip/hip_bf16.h>
#include <math.h>

typedef __attribute__((ext_vector_type(16))) _Float16 v16h;
typedef __attribute__((ext_vector_type(8)))  _Float16 v8h;
typedef __attribute__((ext_vector_type(8)))  float    v8f;

#define D 64

__device__ __forceinline__ float wave_sum(float v) {
#pragma unroll
    for (int m = 16; m > 0; m >>= 1) v += __shfl_xor(v, m, 32);
    return v;
}

// ---------------------------------------------------------------------------
// Init: F0 = ACC = concat(A, B)
// ---------------------------------------------------------------------------
__global__ void init_feats_kernel(const float* __restrict__ A, long nAelem,
                                  const float* __restrict__ Bv, long nBelem,
                                  float* __restrict__ F0, float* __restrict__ ACC) {
    long i = (long)blockIdx.x * blockDim.x + threadIdx.x;
    if (i >= nAelem + nBelem) return;
    float v = (i < nAelem) ? A[i] : Bv[i - nAelem];
    F0[i] = v;
    ACC[i] = v;
}

// ---------------------------------------------------------------------------
// Edge-parallel SpMM: out[rows[e]] += scale * vals[e] * in[cols[e]]
// 16 threads per edge, float4 per thread (D=64). out must be pre-zeroed.
// ---------------------------------------------------------------------------
__global__ void spmm_kernel(const int* __restrict__ rows, const int* __restrict__ cols,
                            const float* __restrict__ vals,
                            const float* __restrict__ in, float* __restrict__ out,
                            int E, float scale) {
    long t = (long)blockIdx.x * blockDim.x + threadIdx.x;
    int e = (int)(t >> 4);
    int l = (int)(t & 15);
    if (e >= E) return;
    float v = vals[e] * scale;
    long c = cols[e], r = rows[e];
    const float4 f = *(const float4*)(in + c * D + l * 4);
    float* o = out + r * D + l * 4;
    atomicAdd(o + 0, v * f.x);
    atomicAdd(o + 1, v * f.y);
    atomicAdd(o + 2, v * f.z);
    atomicAdd(o + 3, v * f.w);
}

// ---------------------------------------------------------------------------
// acc += l2norm(feats), one wave per 64-wide row
// ---------------------------------------------------------------------------
__global__ void acc_l2norm_kernel(float* __restrict__ acc, const float* __restrict__ feats,
                                  int n) {
    int row = blockIdx.x * (blockDim.x >> 5) + (threadIdx.x >> 5);
    if (row >= n) return;
    int lane = threadIdx.x & 31;
    const float2 f = *(const float2*)(feats + (long)row * D + lane * 2);
    float ss = wave_sum(f.x * f.x + f.y * f.y);
    float inv = 1.0f / fmaxf(sqrtf(ss), 1e-12f);
    float* a = acc + (long)row * D + lane * 2;
    a[0] += f.x * inv;
    a[1] += f.y * inv;
}

// ---------------------------------------------------------------------------
// Bundle gather: pred + normalized fp16 copies (pos column) + a_loss term
// One wave per (batch, pair) entry.
// ---------------------------------------------------------------------------
__global__ void gather_bundle_kernel(const float* __restrict__ affb,
                                     const float* __restrict__ acch,
                                     const float* __restrict__ freq,
                                     const int* __restrict__ bundles,
                                     int U, int B,
                                     float* __restrict__ pred,
                                     _Float16* __restrict__ ab16,
                                     _Float16* __restrict__ hb16,
                                     float* __restrict__ aloss_accum) {
    int w = blockIdx.x * (blockDim.x >> 5) + (threadIdx.x >> 5);
    if (w >= 2 * B) return;
    int lane = threadIdx.x & 31;
    int b = w >> 1, p = w & 1;
    int idx = bundles[b * 2 + p];
    const float2 a2 = *(const float2*)(affb + (long)idx * D + lane * 2);
    const float2 h2 = *(const float2*)(acch + ((long)U + idx) * D + lane * 2);
    float sa = wave_sum(a2.x * a2.x + a2.y * a2.y);
    float sh = wave_sum(h2.x * h2.x + h2.y * h2.y);
    float gamma = tanhf(freq[idx]);  // PSI == 1.0
    if (lane == 0) pred[w] = (1.0f - gamma) * sa + gamma * sh;
    if (p == 0) {
        float ia = 1.0f / fmaxf(sqrtf(sa), 1e-12f);
        float ih = 1.0f / fmaxf(sqrtf(sh), 1e-12f);
        long o = (long)b * D + lane * 2;
        ab16[o + 0] = (_Float16)(a2.x * ia);
        ab16[o + 1] = (_Float16)(a2.y * ia);
        hb16[o + 0] = (_Float16)(h2.x * ih);
        hb16[o + 1] = (_Float16)(h2.y * ih);
        float dx = a2.x * ia - h2.x * ih;
        float dy = a2.y * ia - h2.y * ih;
        float al = wave_sum(dx * dx + dy * dy);
        if (lane == 0) atomicAdd(aloss_accum, al);
    }
}

// ---------------------------------------------------------------------------
// User gather: normalized fp16 copies + a_loss term. One wave per batch entry.
// ---------------------------------------------------------------------------
__global__ void gather_user_kernel(const float* __restrict__ acca,
                                   const float* __restrict__ acch,
                                   const int* __restrict__ users,
                                   int B,
                                   _Float16* __restrict__ au16,
                                   _Float16* __restrict__ hu16,
                                   float* __restrict__ aloss_accum) {
    int b = blockIdx.x * (blockDim.x >> 5) + (threadIdx.x >> 5);
    if (b >= B) return;
    int lane = threadIdx.x & 31;
    int u = users[b];
    const float2 a2 = *(const float2*)(acca + (long)u * D + lane * 2);
    const float2 h2 = *(const float2*)(acch + (long)u * D + lane * 2);
    float sa = wave_sum(a2.x * a2.x + a2.y * a2.y);
    float sh = wave_sum(h2.x * h2.x + h2.y * h2.y);
    float ia = 1.0f / fmaxf(sqrtf(sa), 1e-12f);
    float ih = 1.0f / fmaxf(sqrtf(sh), 1e-12f);
    long o = (long)b * D + lane * 2;
    au16[o + 0] = (_Float16)(a2.x * ia);
    au16[o + 1] = (_Float16)(a2.y * ia);
    hu16[o + 0] = (_Float16)(h2.x * ih);
    hu16[o + 1] = (_Float16)(h2.y * ih);
    float dx = a2.x * ia - h2.x * ih;
    float dy = a2.y * ia - h2.y * ih;
    float al = wave_sum(dx * dx + dy * dy);
    if (lane == 0) atomicAdd(aloss_accum, al);
}

// ---------------------------------------------------------------------------
// BPR loss accumulation
// ---------------------------------------------------------------------------
__global__ void bpr_kernel(const float* __restrict__ pred, int B,
                           float* __restrict__ bpr_accum) {
    int b = blockIdx.x * blockDim.x + threadIdx.x;
    if (b >= B) return;
    float z = pred[2 * b] - pred[2 * b + 1];
    float ls = fminf(z, 0.0f) - log1pf(expf(-fabsf(z)));  // log_sigmoid
    atomicAdd(bpr_accum, -ls);
}

// ---------------------------------------------------------------------------
// u_loss Gram reduction via WMMA: G = X * X^T on unit-norm fp16 rows.
// One wave per 16x16 upper-triangular tile; K=64 in two 16x16x32 f16 WMMAs.
// A layout (16-bit A 16x32): lane&15 = row M; lane>>4 selects K half.
// B layout (16-bit B 32x16): lane&15 = col N; lane>>4 selects K half.
// D layout (f32 16x16): N = lane&15, M = r + 8*(lane>>4).
// Epilogue is branchless: exp(-2*d2) <= 1 and the argument is bounded in
// [-8, 0], so the hardware v_exp_f32 (__expf) is exact enough and needs no
// range-reduction/clamp code or EXEC-mask branches.
// ---------------------------------------------------------------------------
__global__ void uloss_gram_kernel(const _Float16* __restrict__ x, int n, int T,
                                  int nTiles, float* __restrict__ accum) {
    int w = blockIdx.x * (blockDim.x >> 5) + (threadIdx.x >> 5);
    if (w >= nTiles) return;
    int lane = threadIdx.x & 31;
    // map linear tile id -> (ti, tj) with tj >= ti (upper triangle, row-major)
    int ti = 0, rem = w;
    while (rem >= T - ti) { rem -= T - ti; ++ti; }
    int tj = ti + rem;

    int half = lane >> 4;
    int l16 = lane & 31 & 15;
    int arow = ti * 16 + l16; if (arow >= n) arow = n - 1;
    int brow = tj * 16 + l16; if (brow >= n) brow = n - 1;
    const _Float16* ap = x + (long)arow * D;
    const _Float16* bp = x + (long)brow * D;

    v8f c = {};
#pragma unroll
    for (int kb = 0; kb < D; kb += 32) {
        v16h a = *(const v16h*)(ap + kb + half * 16);
        v16h b = *(const v16h*)(bp + kb + half * 16);
        c = __builtin_amdgcn_wmma_f32_16x16x32_f16(
            /*neg_a=*/false, a, /*neg_b=*/false, b,
            /*c_mod=*/(short)0, c, /*reuse_a=*/false, /*reuse_b=*/false);
    }

    float local = 0.0f;
    int gj = tj * 16 + l16;
#pragma unroll
    for (int r = 0; r < 8; ++r) {
        int gi = ti * 16 + r + half * 8;
        // rows are unit-norm: d2 = max(1 + 1 - 2*g, 0)
        float d2 = fmaxf(2.0f - 2.0f * c[r], 0.0f);
        float e = __expf(-2.0f * d2);                    // single v_exp_f32
        bool valid = (gj > gi) && (gi < n) && (gj < n);  // strict upper triangle
        local += valid ? e : 0.0f;                       // branchless select
    }
    local = wave_sum(local);
    if (lane == 0) atomicAdd(accum, local);
}

// ---------------------------------------------------------------------------
// Zero small accumulator block
// ---------------------------------------------------------------------------
__global__ void zero_accum_kernel(float* __restrict__ p, int n) {
    int i = threadIdx.x;
    if (i < n) p[i] = 0.0f;
}

// ---------------------------------------------------------------------------
// Final scalar combine
// acc: [0]=bpr_sum [1]=aloss_b [2]=aloss_u [3]=u_ab [4]=u_hb [5]=u_au [6]=u_hu
// ---------------------------------------------------------------------------
__global__ void finalize_kernel(const float* __restrict__ acc, int B,
                                float* __restrict__ out) {
    float fB = (float)B;
    float npairs = 0.5f * fB * (fB - 1.0f);
    out[0] = acc[0] / fB;
    float u_ab = logf(acc[3] / npairs);
    float u_hb = logf(acc[4] / npairs);
    float u_au = logf(acc[5] / npairs);
    float u_hu = logf(acc[6] / npairs);
    float bundle_c = acc[1] / fB + (u_ab + 0.5f * u_hb);
    float user_c   = acc[2] / fB + (u_au + 0.5f * u_hu);
    out[1] = 0.5f * (bundle_c + user_c);
}

// ---------------------------------------------------------------------------
extern "C" void kernel_launch(void* const* d_in, const int* in_sizes, int n_in,
                              void* d_out, int out_size, void* d_ws, size_t ws_size,
                              hipStream_t stream) {
    const float* users_feat   = (const float*)d_in[0];
    const float* items_feat   = (const float*)d_in[1];
    const float* bundles_feat = (const float*)d_in[2];
    const float* bundle_freq  = (const float*)d_in[3];
    const float* aff_vals     = (const float*)d_in[4];
    const float* hist_vals    = (const float*)d_in[5];
    const float* agg_vals     = (const float*)d_in[6];
    const int*   users        = (const int*)d_in[7];
    const int*   bundles      = (const int*)d_in[8];
    const int*   aff_rows     = (const int*)d_in[9];
    const int*   aff_cols     = (const int*)d_in[10];
    const int*   hist_rows    = (const int*)d_in[11];
    const int*   hist_cols    = (const int*)d_in[12];
    const int*   agg_rows     = (const int*)d_in[13];
    const int*   agg_cols     = (const int*)d_in[14];

    const int U  = in_sizes[0] / D;
    const int I  = in_sizes[1] / D;
    const int Bn = in_sizes[2] / D;
    const int E_aff  = in_sizes[4];
    const int E_hist = in_sizes[5];
    const int E_agg  = in_sizes[6];
    const int B      = in_sizes[7];   // users is (B,1)
    const int n1 = U + I;
    const int n2 = U + Bn;

    // ---- workspace carving (256B aligned) ----
    char* ws = (char*)d_ws;
    size_t off = 0;
    auto carve = [&](size_t bytes) -> char* {
        char* p = ws + off;
        off = (off + bytes + 255) & ~(size_t)255;
        return p;
    };
    float*     F0    = (float*)carve((size_t)n1 * D * sizeof(float));
    float*     F1    = (float*)carve((size_t)n1 * D * sizeof(float));
    float*     ACC_A = (float*)carve((size_t)n1 * D * sizeof(float));
    float*     ACC_H = (float*)carve((size_t)n2 * D * sizeof(float));
    float*     AFFB  = (float*)carve((size_t)Bn * D * sizeof(float));
    _Float16*  AB16  = (_Float16*)carve((size_t)B * D * sizeof(_Float16));
    _Float16*  HB16  = (_Float16*)carve((size_t)B * D * sizeof(_Float16));
    _Float16*  AU16  = (_Float16*)carve((size_t)B * D * sizeof(_Float16));
    _Float16*  HU16  = (_Float16*)carve((size_t)B * D * sizeof(_Float16));
    float*     PRED  = (float*)carve((size_t)B * 2 * sizeof(float));
    float*     ACCUM = (float*)carve(64);

    zero_accum_kernel<<<1, 32, 0, stream>>>(ACCUM, 8);

    const int TPB = 256;
    auto blocks_for = [](long threads) { return (int)((threads + 255) / 256); };

    // ================= aff graph propagation (n1 nodes) =================
    init_feats_kernel<<<blocks_for((long)n1 * D), TPB, 0, stream>>>(
        users_feat, (long)U * D, items_feat, (long)I * D, F0, ACC_A);
    hipMemsetAsync(F1, 0, (size_t)n1 * D * sizeof(float), stream);
    spmm_kernel<<<blocks_for((long)E_aff * 16), TPB, 0, stream>>>(
        aff_rows, aff_cols, aff_vals, F0, F1, E_aff, 1.0f / 2.0f);
    acc_l2norm_kernel<<<(n1 + 7) / 8, TPB, 0, stream>>>(ACC_A, F1, n1);
    hipMemsetAsync(F0, 0, (size_t)n1 * D * sizeof(float), stream);
    spmm_kernel<<<blocks_for((long)E_aff * 16), TPB, 0, stream>>>(
        aff_rows, aff_cols, aff_vals, F1, F0, E_aff, 1.0f / 3.0f);
    acc_l2norm_kernel<<<(n1 + 7) / 8, TPB, 0, stream>>>(ACC_A, F0, n1);

    // ================= hist graph propagation (n2 nodes) =================
    init_feats_kernel<<<blocks_for((long)n2 * D), TPB, 0, stream>>>(
        users_feat, (long)U * D, bundles_feat, (long)Bn * D, F0, ACC_H);
    hipMemsetAsync(F1, 0, (size_t)n2 * D * sizeof(float), stream);
    spmm_kernel<<<blocks_for((long)E_hist * 16), TPB, 0, stream>>>(
        hist_rows, hist_cols, hist_vals, F0, F1, E_hist, 1.0f / 2.0f);
    acc_l2norm_kernel<<<(n2 + 7) / 8, TPB, 0, stream>>>(ACC_H, F1, n2);
    hipMemsetAsync(F0, 0, (size_t)n2 * D * sizeof(float), stream);
    spmm_kernel<<<blocks_for((long)E_hist * 16), TPB, 0, stream>>>(
        hist_rows, hist_cols, hist_vals, F1, F0, E_hist, 1.0f / 3.0f);
    acc_l2norm_kernel<<<(n2 + 7) / 8, TPB, 0, stream>>>(ACC_H, F0, n2);

    // ================= aggregation: aff_b = agg @ aff_i =================
    hipMemsetAsync(AFFB, 0, (size_t)Bn * D * sizeof(float), stream);
    spmm_kernel<<<blocks_for((long)E_agg * 16), TPB, 0, stream>>>(
        agg_rows, agg_cols, agg_vals, ACC_A + (long)U * D, AFFB, E_agg, 1.0f);

    // ================= batch gathers =================
    gather_bundle_kernel<<<(2 * B + 7) / 8, TPB, 0, stream>>>(
        AFFB, ACC_H, bundle_freq, bundles, U, B, PRED, AB16, HB16, ACCUM + 1);
    gather_user_kernel<<<(B + 7) / 8, TPB, 0, stream>>>(
        ACC_A, ACC_H, users, B, AU16, HU16, ACCUM + 2);

    // ================= BPR loss =================
    bpr_kernel<<<blocks_for(B), TPB, 0, stream>>>(PRED, B, ACCUM + 0);

    // ================= u_loss Gram reductions (WMMA) =================
    const int T = (B + 15) / 16;
    const int nTiles = T * (T + 1) / 2;
    const int ublocks = (nTiles + 7) / 8;
    uloss_gram_kernel<<<ublocks, TPB, 0, stream>>>(AB16, B, T, nTiles, ACCUM + 3);
    uloss_gram_kernel<<<ublocks, TPB, 0, stream>>>(HB16, B, T, nTiles, ACCUM + 4);
    uloss_gram_kernel<<<ublocks, TPB, 0, stream>>>(AU16, B, T, nTiles, ACCUM + 5);
    uloss_gram_kernel<<<ublocks, TPB, 0, stream>>>(HU16, B, T, nTiles, ACCUM + 6);

    // ================= finalize =================
    finalize_kernel<<<1, 1, 0, stream>>>(ACCUM, B, (float*)d_out);
}
